// LoSDecoderWithAttention_26585847562812
// MI455X (gfx1250) — matrix operand order repaired and checked by reading, not compiled, not run on hardware
//
#include <hip/hip_runtime.h>

// ---------------------------------------------------------------------------
// Problem constants
// ---------------------------------------------------------------------------
#define BB   1024
#define TT   100
#define EE   512
#define DD   1024
#define ENCC 1024
#define VV   32000
#define KX   2560          // E + ENC + D  (concat GEMM K for gates)

// d_out layout (floats): preds | los_s | dec_len | alphas | sort_ind
static const size_t OFF_PRED    = 0;
static const size_t OFF_LOS     = (size_t)BB * TT;                       // 102400
static const size_t OFF_DECLEN  = OFF_LOS + BB;                          // 103424
static const size_t OFF_ALPHA   = OFF_DECLEN + BB;                       // 104448
static const size_t OFF_SORTIND = OFF_ALPHA + (size_t)BB * TT * ENCC;    // +104857600

// ---------------------------------------------------------------------------
// Types for WMMA
// ---------------------------------------------------------------------------
typedef __attribute__((ext_vector_type(16))) __bf16 v16bf;
typedef __attribute__((ext_vector_type(8)))  __bf16 v8bf;
typedef __attribute__((ext_vector_type(8)))  float  v8f;

__device__ __forceinline__ unsigned short f2bf(float f) {
    unsigned u = __float_as_uint(f);
    unsigned r = u + 0x7FFFu + ((u >> 16) & 1u);   // round-to-nearest-even
    return (unsigned short)(r >> 16);
}
__device__ __forceinline__ float sigmoidf_(float x) {
    return 1.f / (1.f + __expf(-x));
}

// ---------------------------------------------------------------------------
// BF16 WMMA GEMM:  C[M=1024, N] = A[M,K](bf16, row stride lda) @ W[N,K]^T + bias
// grid = (N/256, 1024/64), block = 256 (8 waves, 2Mx4N).
// Each wave: 32x64 output = 2x4 tiles of 16x16; B frag reused across M tiles,
// A frags reused across the 4 N tiles -> 12 b128 loads per 8 WMMAs.
// ---------------------------------------------------------------------------
__global__ __launch_bounds__(256)
void gemm_bf16_wmma(const unsigned short* __restrict__ Ap, int lda,
                    const unsigned short* __restrict__ Wp,
                    const float* __restrict__ bias,
                    float* __restrict__ C, int N, int K)
{
    const int lane = threadIdx.x & 31;
    const int wave = threadIdx.x >> 5;
    const int wm   = wave >> 2;              // 0..1
    const int wn   = wave & 3;               // 0..3
    const int m0   = blockIdx.y * 64  + wm * 32;
    const int n0   = blockIdx.x * 256 + wn * 64;
    const int lm   = lane & 15;
    const int hi   = lane >> 4;              // 0 or 1

    const __bf16* A = reinterpret_cast<const __bf16*>(Ap);
    const __bf16* W = reinterpret_cast<const __bf16*>(Wp);

    v8f acc[2][4];
#pragma unroll
    for (int mt = 0; mt < 2; ++mt)
#pragma unroll
        for (int t = 0; t < 4; ++t)
            acc[mt][t] = v8f{0.f,0.f,0.f,0.f,0.f,0.f,0.f,0.f};

    const __bf16* aRow0 = A + (size_t)(m0 + lm) * lda;
    const __bf16* aRow1 = A + (size_t)(m0 + 16 + lm) * lda;
    const int aOff = hi * 8;                 // A: lanes 0-15 -> K 0..7 / 16..23, lanes 16-31 -> 8..15 / 24..31
    const int bOff = hi * 8;                 // B: lanes 0-15 -> K 0..7 / 16..23, lanes 16-31 -> 8..15 / 24..31

    for (int k0 = 0; k0 < K; k0 += 32) {
        __builtin_prefetch(aRow0 + k0 + 64, 0, 0);   // global_prefetch_b8 on the A stream

        v8bf a0lo = *reinterpret_cast<const v8bf*>(aRow0 + k0 + aOff);
        v8bf a0hi = *reinterpret_cast<const v8bf*>(aRow0 + k0 + aOff + 16);
        v8bf a1lo = *reinterpret_cast<const v8bf*>(aRow1 + k0 + aOff);
        v8bf a1hi = *reinterpret_cast<const v8bf*>(aRow1 + k0 + aOff + 16);
        v16bf a0, a1;
#pragma unroll
        for (int i = 0; i < 8; ++i) {
            a0[i] = a0lo[i]; a0[8 + i] = a0hi[i];
            a1[i] = a1lo[i]; a1[8 + i] = a1hi[i];
        }

#pragma unroll
        for (int t = 0; t < 4; ++t) {
            const __bf16* wRow = W + (size_t)(n0 + t * 16 + lm) * K + k0 + bOff;
            v8bf blo = *reinterpret_cast<const v8bf*>(wRow);
            v8bf bhi = *reinterpret_cast<const v8bf*>(wRow + 16);
            v16bf b;
#pragma unroll
            for (int i = 0; i < 8; ++i) { b[i] = blo[i]; b[8 + i] = bhi[i]; }
            acc[0][t] = __builtin_amdgcn_wmma_f32_16x16x32_bf16(
                false, a0, false, b, (short)0, acc[0][t], false, false);
            acc[1][t] = __builtin_amdgcn_wmma_f32_16x16x32_bf16(
                false, a1, false, b, (short)0, acc[1][t], false, false);
        }
    }

#pragma unroll
    for (int mt = 0; mt < 2; ++mt) {
#pragma unroll
        for (int t = 0; t < 4; ++t) {
            const int col = n0 + t * 16 + lm;
            const float bv = bias[col];
#pragma unroll
            for (int r = 0; r < 8; ++r) {
                const int row = m0 + mt * 16 + r + hi * 8;
                C[(size_t)row * N + col] = acc[mt][t][r] + bv;
            }
        }
    }
}

// ---------------------------------------------------------------------------
// Stable descending argsort of pathway_len (bitonic, one block of 1024).
// Writes los_s, dec_len, sort_ind outputs + int scratch copies.
// ---------------------------------------------------------------------------
__global__ __launch_bounds__(1024)
void sort_setup_kernel(const int* __restrict__ pl, const float* __restrict__ los,
                       int* __restrict__ sortind, int* __restrict__ declen,
                       float* __restrict__ out)
{
    __shared__ unsigned key[1024];
    const int tid = threadIdx.x;
    unsigned p = (unsigned)pl[tid];                 // (B,1) flat
    key[tid] = (p << 10) | (1023u - (unsigned)tid); // stable tie-break: lower idx first
    __syncthreads();
    for (int k = 2; k <= 1024; k <<= 1) {
        for (int j = k >> 1; j > 0; j >>= 1) {
            int ixj = tid ^ j;
            if (ixj > tid) {
                bool up = ((tid & k) == 0);
                unsigned a = key[tid], b = key[ixj];
                bool sw = up ? (a < b) : (a > b);   // descending overall
                if (sw) { key[tid] = b; key[ixj] = a; }
            }
            __syncthreads();
        }
    }
    unsigned kk = key[tid];
    int idx  = 1023 - (int)(kk & 1023u);
    int plen = (int)(kk >> 10);
    sortind[tid] = idx;
    declen[tid]  = plen - 1;
    out[OFF_LOS + tid]     = los[idx];
    out[OFF_DECLEN + tid]  = (float)(plen - 1);
    out[OFF_SORTIND + tid] = (float)idx;
}

// ---------------------------------------------------------------------------
// Gather sorted encoder rows (f32 + bf16 copies).
// ---------------------------------------------------------------------------
__global__ __launch_bounds__(256)
void enc_gather_kernel(const float* __restrict__ enc, const int* __restrict__ sortind,
                       float* __restrict__ encS, unsigned short* __restrict__ encB)
{
    const int b = blockIdx.x, tid = threadIdx.x;
    const float* r = enc + (size_t)sortind[b] * ENCC;
#pragma unroll
    for (int i = 0; i < 4; ++i) {
        int j = tid + i * 256;
        float v = r[j];
        encS[(size_t)b * ENCC + j] = v;
        encB[(size_t)b * ENCC + j] = f2bf(v);
    }
}

// ---------------------------------------------------------------------------
// Weight repack kernels (f32 -> bf16, concatenated)
// ---------------------------------------------------------------------------
__global__ void pack_rows2(const float* __restrict__ A, const float* __restrict__ B,
                           unsigned short* __restrict__ out, int rows, int K)
{
    size_t total = (size_t)(2 * rows) * K;
    for (size_t i = (size_t)blockIdx.x * blockDim.x + threadIdx.x; i < total;
         i += (size_t)gridDim.x * blockDim.x) {
        int n = (int)(i / K), k = (int)(i % K);
        float v = (n < rows) ? A[(size_t)n * K + k] : B[(size_t)(n - rows) * K + k];
        out[i] = f2bf(v);
    }
}
__global__ void pack_ih_hh(const float* __restrict__ Wih, const float* __restrict__ Whh,
                           unsigned short* __restrict__ out)
{
    size_t total = (size_t)(4 * DD) * KX;
    for (size_t i = (size_t)blockIdx.x * blockDim.x + threadIdx.x; i < total;
         i += (size_t)gridDim.x * blockDim.x) {
        int n = (int)(i / KX), k = (int)(i % KX);
        float v = (k < EE + ENCC) ? Wih[(size_t)n * (EE + ENCC) + k]
                                  : Whh[(size_t)n * DD + (k - (EE + ENCC))];
        out[i] = f2bf(v);
    }
}
__global__ void bias_concat(const float* __restrict__ a, const float* __restrict__ b,
                            float* __restrict__ out, int n1, int n2)
{
    int i = blockIdx.x * blockDim.x + threadIdx.x;
    if (i < n1) out[i] = a[i];
    else if (i < n1 + n2) out[i] = b[i - n1];
}
__global__ void bias_add(const float* __restrict__ a, const float* __restrict__ b,
                         float* __restrict__ out, int n)
{
    int i = blockIdx.x * blockDim.x + threadIdx.x;
    if (i < n) out[i] = a[i] + b[i];
}

// ---------------------------------------------------------------------------
// Split [h0|c0] GEMM output into h, c (f32) + bf16 h-region of x buffer.
// ---------------------------------------------------------------------------
__global__ __launch_bounds__(256)
void init_split_kernel(const float* __restrict__ C0, float* __restrict__ h,
                       float* __restrict__ c, unsigned short* __restrict__ xbuf)
{
    const int b = blockIdx.x, tid = threadIdx.x;
    const float* r = C0 + (size_t)b * 2048;
#pragma unroll
    for (int i = 0; i < 4; ++i) {
        int j = tid + i * 256;
        float hv = r[j], cv = r[DD + j];
        h[(size_t)b * DD + j] = hv;
        c[(size_t)b * DD + j] = cv;
        xbuf[(size_t)b * KX + (EE + ENCC) + j] = f2bf(hv);
    }
}

// ---------------------------------------------------------------------------
// Gather embedding row for timestep t into x buffer (bf16).
// ---------------------------------------------------------------------------
__global__ __launch_bounds__(128)
void emb_gather_kernel(const float* __restrict__ emb, const int* __restrict__ paths,
                       const int* __restrict__ sortind, unsigned short* __restrict__ xbuf,
                       int t)
{
    const int b = blockIdx.x, tid = threadIdx.x;
    const int token = paths[(size_t)sortind[b] * TT + t];
    const float* er = emb + (size_t)token * EE;
    unsigned short* xr = xbuf + (size_t)b * KX;
#pragma unroll
    for (int i = 0; i < 4; ++i) {
        int j = tid + i * 128;
        xr[j] = f2bf(er[j]);
    }
}

// ---------------------------------------------------------------------------
// att = relu(enc + h@W_attT + b_att) ; alpha = softmax(att) ;
// gate = sigmoid(h@W_fbetaT + b_fbeta) ; awe = gate*enc*alpha -> x buffer
// alphas_out[:,t,:] = active ? alpha : 0
// ---------------------------------------------------------------------------
__global__ __launch_bounds__(256)
void attn_softmax_kernel(const float* __restrict__ encS, const float* __restrict__ C1,
                         const int* __restrict__ declen,
                         unsigned short* __restrict__ xbuf,
                         float* __restrict__ alphaOut, int t)
{
    __shared__ float red[256];
    const int b = blockIdx.x, tid = threadIdx.x;
    const bool active = t < declen[b];
    const float* e  = encS + (size_t)b * ENCC;
    const float* cr = C1   + (size_t)b * 2048;

    float av[4];
    float m = -3.4e38f;
#pragma unroll
    for (int i = 0; i < 4; ++i) {
        int j = tid + i * 256;
        float v = e[j] + cr[j];
        v = v > 0.f ? v : 0.f;
        av[i] = v;
        m = fmaxf(m, v);
    }
    red[tid] = m; __syncthreads();
    for (int s = 128; s > 0; s >>= 1) {
        if (tid < s) red[tid] = fmaxf(red[tid], red[tid + s]);
        __syncthreads();
    }
    m = red[0]; __syncthreads();

    float sum = 0.f;
#pragma unroll
    for (int i = 0; i < 4; ++i) { av[i] = __expf(av[i] - m); sum += av[i]; }
    red[tid] = sum; __syncthreads();
    for (int s = 128; s > 0; s >>= 1) {
        if (tid < s) red[tid] += red[tid + s];
        __syncthreads();
    }
    const float inv = 1.f / red[0];

    float* aout = alphaOut + ((size_t)b * TT + t) * ENCC;
    unsigned short* xr = xbuf + (size_t)b * KX + EE;
#pragma unroll
    for (int i = 0; i < 4; ++i) {
        int j = tid + i * 256;
        float alpha = av[i] * inv;
        float gate  = sigmoidf_(cr[ENCC + j]);
        float awe   = gate * e[j] * alpha;
        xr[j]  = f2bf(awe);
        aout[j] = active ? alpha : 0.f;
    }
}

// ---------------------------------------------------------------------------
// LSTM pointwise update + prediction reduction.
// ---------------------------------------------------------------------------
__global__ __launch_bounds__(256)
void lstm_kernel(const float* __restrict__ C2, float* __restrict__ h, float* __restrict__ c,
                 unsigned short* __restrict__ xbuf,
                 const float* __restrict__ Wfc, const float* __restrict__ bfc,
                 const int* __restrict__ declen, float* __restrict__ predOut, int t)
{
    __shared__ float red[256];
    const int b = blockIdx.x, tid = threadIdx.x;
    const bool active = t < declen[b];
    const float* g = C2 + (size_t)b * (4 * DD);
    float* hb = h + (size_t)b * DD;
    float* cb = c + (size_t)b * DD;
    unsigned short* xh = xbuf + (size_t)b * KX + (EE + ENCC);

    float psum = 0.f;
#pragma unroll
    for (int i = 0; i < 4; ++i) {
        int j = tid + i * 256;
        float ig = sigmoidf_(g[j]);
        float fg = sigmoidf_(g[DD + j]);
        float gg = tanhf(g[2 * DD + j]);
        float og = sigmoidf_(g[3 * DD + j]);
        float cn = fg * cb[j] + ig * gg;
        float hn = og * tanhf(cn);
        psum += hn * Wfc[j];
        float hk = active ? hn : hb[j];
        float ck = active ? cn : cb[j];
        hb[j] = hk; cb[j] = ck;
        xh[j] = f2bf(hk);
    }
    red[tid] = psum; __syncthreads();
    for (int s = 128; s > 0; s >>= 1) {
        if (tid < s) red[tid] += red[tid + s];
        __syncthreads();
    }
    if (tid == 0) predOut[(size_t)b * TT + t] = active ? (red[0] + bfc[0]) : 0.f;
}

// ---------------------------------------------------------------------------
// Host orchestration
// ---------------------------------------------------------------------------
extern "C" void kernel_launch(void* const* d_in, const int* in_sizes, int n_in,
                              void* d_out, int out_size, void* d_ws, size_t ws_size,
                              hipStream_t stream)
{
    const float* encoder_out = (const float*)d_in[0];
    const int*   paths       = (const int*)  d_in[1];
    const int*   pathway_len = (const int*)  d_in[2];
    const float* los         = (const float*)d_in[3];
    const float* emb         = (const float*)d_in[4];
    const float* W_att       = (const float*)d_in[5];
    const float* b_att       = (const float*)d_in[6];
    const float* W_init_h    = (const float*)d_in[7];
    const float* b_init_h    = (const float*)d_in[8];
    const float* W_init_c    = (const float*)d_in[9];
    const float* b_init_c    = (const float*)d_in[10];
    const float* W_fbeta     = (const float*)d_in[11];
    const float* b_fbeta     = (const float*)d_in[12];
    const float* W_ih        = (const float*)d_in[13];
    const float* b_ih        = (const float*)d_in[14];
    const float* W_hh        = (const float*)d_in[15];
    const float* b_hh        = (const float*)d_in[16];
    const float* W_fc        = (const float*)d_in[17];
    const float* b_fc        = (const float*)d_in[18];
    float* out = (float*)d_out;

    // ---- workspace carve-up (256B aligned) ----
    size_t off = 0;
    auto alloc = [&](size_t bytes) -> void* {
        off = (off + 255) & ~(size_t)255;
        void* p = (char*)d_ws + off;
        off += bytes;
        return p;
    };
    int*            sortind = (int*)alloc((size_t)BB * 4);
    int*            declen  = (int*)alloc((size_t)BB * 4);
    float*          encS    = (float*)alloc((size_t)BB * ENCC * 4);
    unsigned short* encB    = (unsigned short*)alloc((size_t)BB * ENCC * 2);
    unsigned short* Wattg   = (unsigned short*)alloc((size_t)2048 * 1024 * 2);
    float*          battg   = (float*)alloc(2048 * 4);
    unsigned short* Wihhh   = (unsigned short*)alloc((size_t)4096 * KX * 2);
    float*          bihhh   = (float*)alloc(4096 * 4);
    unsigned short* Winit   = (unsigned short*)alloc((size_t)2048 * 1024 * 2);
    float*          binit   = (float*)alloc(2048 * 4);
    float*          C1      = (float*)alloc((size_t)BB * 2048 * 4);  // attgate out / init out
    float*          C2      = (float*)alloc((size_t)BB * 4096 * 4);  // gates out
    unsigned short* xbuf    = (unsigned short*)alloc((size_t)BB * KX * 2);
    float*          hbuf    = (float*)alloc((size_t)BB * DD * 4);
    float*          cbuf    = (float*)alloc((size_t)BB * DD * 4);
    (void)ws_size; (void)n_in; (void)in_sizes; (void)out_size;

    // ---- setup ----
    sort_setup_kernel<<<1, 1024, 0, stream>>>(pathway_len, los, sortind, declen, out);
    enc_gather_kernel<<<BB, 256, 0, stream>>>(encoder_out, sortind, encS, encB);

    pack_rows2<<<2048, 256, 0, stream>>>(W_att, W_fbeta, Wattg, 1024, 1024);
    bias_concat<<<8, 256, 0, stream>>>(b_att, b_fbeta, battg, ENCC, ENCC);
    pack_rows2<<<2048, 256, 0, stream>>>(W_init_h, W_init_c, Winit, 1024, 1024);
    bias_concat<<<8, 256, 0, stream>>>(b_init_h, b_init_c, binit, DD, DD);
    pack_ih_hh<<<4096, 256, 0, stream>>>(W_ih, W_hh, Wihhh);
    bias_add<<<16, 256, 0, stream>>>(b_ih, b_hh, bihhh, 4096);

    // h0|c0 = enc_bf16 @ Winit^T + binit   (M=1024, N=2048, K=1024)
    gemm_bf16_wmma<<<dim3(2048 / 256, BB / 64), 256, 0, stream>>>(
        encB, ENCC, Winit, binit, C1, 2048, 1024);
    init_split_kernel<<<BB, 256, 0, stream>>>(C1, hbuf, cbuf, xbuf);

    // ---- decode loop ----
    const unsigned short* hbf = xbuf + (EE + ENCC);  // bf16 h lives inside x, lda = KX
    for (int t = 0; t < TT; ++t) {
        emb_gather_kernel<<<BB, 128, 0, stream>>>(emb, paths, sortind, xbuf, t);
        // [attpre | gatepre] = h @ [W_att;W_fbeta]^T + [b_att;b_fbeta]
        gemm_bf16_wmma<<<dim3(2048 / 256, BB / 64), 256, 0, stream>>>(
            hbf, KX, Wattg, battg, C1, 2048, 1024);
        attn_softmax_kernel<<<BB, 256, 0, stream>>>(
            encS, C1, declen, xbuf, out + OFF_ALPHA, t);
        // gates = [emb|awe|h] @ [W_ih|W_hh]^T + (b_ih+b_hh)
        gemm_bf16_wmma<<<dim3(4096 / 256, BB / 64), 256, 0, stream>>>(
            xbuf, KX, Wihhh, bihhh, C2, 4096, KX);
        lstm_kernel<<<BB, 256, 0, stream>>>(
            C2, hbuf, cbuf, xbuf, W_fc, b_fc, declen, out + OFF_PRED, t);
    }
}
